// UlyssesAttention_50766513439339
// MI455X (gfx1250) — matrix operand order repaired and checked by reading, not compiled, not run on hardware
//
#include <hip/hip_runtime.h>

#define BQ 2
#define SQ 4096
#define HQ 1024

typedef __attribute__((ext_vector_type(16))) __bf16 bf16x16;
typedef __attribute__((ext_vector_type(8)))  __bf16 bf16x8;
typedef __attribute__((ext_vector_type(8)))  float  f32x8;

union AFrag {
    bf16x16 v;
    struct { bf16x8 lo; bf16x8 hi; } p;
};

__device__ __forceinline__ f32x8 wmma_bf16(bf16x16 a, bf16x16 b, f32x8 c) {
    // D = A(16x32 bf16) * B(32x16 bf16) + C(16x16 f32)
    return __builtin_amdgcn_wmma_f32_16x16x32_bf16(
        /*neg_a=*/false, a, /*neg_b=*/false, b,
        /*c_mod=*/(short)0, c, /*reuse_a=*/false, /*reuse_b=*/false);
}

__device__ __forceinline__ f32x8 zero8() {
    f32x8 z = {0.f,0.f,0.f,0.f,0.f,0.f,0.f,0.f};
    return z;
}

// ---------------------------------------------------------------- convert
__global__ __launch_bounds__(256) void cvt_f32_to_bf16(const float* __restrict__ in,
                                                       __bf16* __restrict__ out, int n) {
    int i = blockIdx.x * 256 + threadIdx.x;
    if (i < n) out[i] = (__bf16)in[i];
}

// ---------------------------------------------------------------- QKV GEMM
// Each wave computes a 32(M) x 64(N) block: 2 A-frags x 4 B-frags -> 8 WMMAs
// per K-step, ~21 FLOP/byte from L2.
// q,k stored bf16 row-major [B*S, H]; v stored transposed bf16 [B, H, S]
__global__ __launch_bounds__(256) void qkv_gemm_kernel(const __bf16* __restrict__ xb,
                                                       const __bf16* __restrict__ wq,
                                                       const float*  __restrict__ bias,
                                                       __bf16* __restrict__ qb,
                                                       __bf16* __restrict__ kb,
                                                       __bf16* __restrict__ vtb) {
    const int wave = threadIdx.x >> 5;
    const int lane = threadIdx.x & 31;
    const int half = lane >> 4;
    const int mr   = lane & 15;
    const int m0 = blockIdx.x * 32;
    const int n0 = (blockIdx.y * 8 + wave) * 64;

    const __bf16* arow0 = xb + (size_t)(m0 + mr) * HQ;
    const __bf16* arow1 = xb + (size_t)(m0 + 16 + mr) * HQ;
    const __bf16* brow  = wq + (size_t)(n0 + mr) * HQ;

    f32x8 acc[2][4];
#pragma unroll
    for (int mi = 0; mi < 2; ++mi)
#pragma unroll
        for (int t = 0; t < 4; ++t) acc[mi][t] = zero8();

    for (int k0 = 0; k0 < HQ; k0 += 32) {
        AFrag a0, a1;
        a0.p.lo = *(const bf16x8*)(arow0 + k0 + 16 * half);
        a0.p.hi = *(const bf16x8*)(arow0 + k0 + 16 * half + 8);
        a1.p.lo = *(const bf16x8*)(arow1 + k0 + 16 * half);
        a1.p.hi = *(const bf16x8*)(arow1 + k0 + 16 * half + 8);
#pragma unroll
        for (int t = 0; t < 4; ++t) {
            bf16x16 bfr = *(const bf16x16*)(brow + (size_t)t * 16 * HQ + k0 + 16 * half);
            acc[0][t] = wmma_bf16(a0.v, bfr, acc[0][t]);
            acc[1][t] = wmma_bf16(a1.v, bfr, acc[1][t]);
        }
    }

#pragma unroll
    for (int t = 0; t < 4; ++t) {
        const int n = n0 + t * 16 + mr;
        const float bv = bias[n];
#pragma unroll
        for (int mi = 0; mi < 2; ++mi) {
#pragma unroll
            for (int r = 0; r < 8; ++r) {
                const int m = m0 + 16 * mi + 8 * half + r;
                const float v = acc[mi][t][r] + bv;
                if (n < HQ) {
                    qb[(size_t)m * HQ + n] = (__bf16)v;
                } else if (n < 2 * HQ) {
                    kb[(size_t)m * HQ + (n - HQ)] = (__bf16)v;
                } else {
                    const int bidx = m / SQ, s = m - bidx * SQ;
                    vtb[((size_t)bidx * HQ + (n - 2 * HQ)) * SQ + s] = (__bf16)v;
                }
            }
        }
    }
}

// ---------------------------------------------------------------- attention
// One workgroup (8 waves) per 16-row query tile. LDS: 16x4096 f32 scores
// (256 KB, CDNA5-only) + 16x1024 bf16 staged q tile (32 KB).
__global__ __launch_bounds__(256) void attn_kernel(const __bf16* __restrict__ qb,
                                                   const __bf16* __restrict__ kb,
                                                   const __bf16* __restrict__ vtb,
                                                   __bf16* __restrict__ ctxb) {
    extern __shared__ char smem_raw[];
    float*  sc = (float*)smem_raw;                       // [16][SQ]
    __bf16* qt = (__bf16*)(smem_raw + 16 * SQ * 4);      // [16][HQ]

    const int wave = threadIdx.x >> 5;
    const int lane = threadIdx.x & 31;
    const int half = lane >> 4;
    const int mr   = lane & 15;
    const int b  = blockIdx.x / (SQ / 16);
    const int q0 = (blockIdx.x % (SQ / 16)) * 16;

    // phase 0: stage q tile (32 KB) via CDNA5 async global->LDS copies.
    // 2048 x 16B chunks over 256 threads = 8 chunks/thread.
    {
        const char* gsrc = (const char*)(qb + ((size_t)b * SQ + q0) * HQ);
        const unsigned qt_lds = (unsigned)(unsigned long long)(void*)qt;
#pragma unroll
        for (int i = 0; i < 8; ++i) {
            const unsigned byte_off = (threadIdx.x + i * 256) * 16u;
            unsigned lds_addr = qt_lds + byte_off;
            unsigned long long gaddr = (unsigned long long)(gsrc + byte_off);
            asm volatile("global_load_async_to_lds_b128 %0, %1, off"
                         :: "v"(lds_addr), "v"(gaddr) : "memory");
        }
        asm volatile("s_wait_asynccnt 0x0" ::: "memory");
    }
    __syncthreads();

    // phase 1: scores = q @ k^T * (1/sqrt(H)) -> LDS f32
    const __bf16* kbb = kb + (size_t)b * SQ * HQ;
    for (int g = 0; g < 4; ++g) {
        const int jt0 = wave * 32 + g * 8;
        f32x8 acc[8];
#pragma unroll
        for (int t = 0; t < 8; ++t) acc[t] = zero8();
        for (int ks = 0; ks < HQ; ks += 32) {
            AFrag a;
            a.p.lo = *(const bf16x8*)(qt + mr * HQ + ks + 16 * half);
            a.p.hi = *(const bf16x8*)(qt + mr * HQ + ks + 16 * half + 8);
#pragma unroll
            for (int t = 0; t < 8; ++t) {
                bf16x16 bfr = *(const bf16x16*)(kbb + (size_t)((jt0 + t) * 16 + mr) * HQ
                                                + ks + 16 * half);
                acc[t] = wmma_bf16(a.v, bfr, acc[t]);
            }
        }
#pragma unroll
        for (int t = 0; t < 8; ++t)
#pragma unroll
            for (int r = 0; r < 8; ++r)
                sc[(size_t)(8 * half + r) * SQ + (jt0 + t) * 16 + mr] = acc[t][r] * 0.03125f;
    }
    __syncthreads();

    // phase 2: row softmax (2 rows per wave), wave32 shuffle reductions
    for (int rr = 0; rr < 2; ++rr) {
        const int row = wave * 2 + rr;
        float* srow = sc + (size_t)row * SQ;
        float mx = -3.0e38f;
        for (int j = lane; j < SQ; j += 32) mx = fmaxf(mx, srow[j]);
#pragma unroll
        for (int o = 16; o >= 1; o >>= 1) mx = fmaxf(mx, __shfl_xor(mx, o, 32));
        float sum = 0.f;
        for (int j = lane; j < SQ; j += 32) {
            float e = __expf(srow[j] - mx);
            srow[j] = e;
            sum += e;
        }
#pragma unroll
        for (int o = 16; o >= 1; o >>= 1) sum += __shfl_xor(sum, o, 32);
        const float inv = 1.0f / sum;
        for (int j = lane; j < SQ; j += 32) srow[j] *= inv;
    }
    __syncthreads();

    // phase 3: ctx = P @ v   (v stored transposed [H][S] -> contiguous B frags)
    const __bf16* vbb = vtb + (size_t)b * HQ * SQ;
    {
        const int nt0 = wave * 8;
        f32x8 acc[8];
#pragma unroll
        for (int t = 0; t < 8; ++t) acc[t] = zero8();
        for (int ks = 0; ks < SQ; ks += 32) {
            const float* prow = sc + (size_t)mr * SQ + ks;
            float4 f0 = *(const float4*)(prow + 16 * half);
            float4 f1 = *(const float4*)(prow + 16 * half + 4);
            float4 f2 = *(const float4*)(prow + 16 * half + 8);
            float4 f3 = *(const float4*)(prow + 16 * half + 12);
            AFrag a;
            a.v[0]  = (__bf16)f0.x; a.v[1]  = (__bf16)f0.y;
            a.v[2]  = (__bf16)f0.z; a.v[3]  = (__bf16)f0.w;
            a.v[4]  = (__bf16)f1.x; a.v[5]  = (__bf16)f1.y;
            a.v[6]  = (__bf16)f1.z; a.v[7]  = (__bf16)f1.w;
            a.v[8]  = (__bf16)f2.x; a.v[9]  = (__bf16)f2.y;
            a.v[10] = (__bf16)f2.z; a.v[11] = (__bf16)f2.w;
            a.v[12] = (__bf16)f3.x; a.v[13] = (__bf16)f3.y;
            a.v[14] = (__bf16)f3.z; a.v[15] = (__bf16)f3.w;
#pragma unroll
            for (int t = 0; t < 8; ++t) {
                bf16x16 bfr = *(const bf16x16*)(vbb + (size_t)((nt0 + t) * 16 + mr) * SQ
                                                + ks + 16 * half);
                acc[t] = wmma_bf16(a.v, bfr, acc[t]);
            }
        }
#pragma unroll
        for (int t = 0; t < 8; ++t)
#pragma unroll
            for (int r = 0; r < 8; ++r)
                ctxb[((size_t)b * SQ + q0 + 8 * half + r) * HQ + (nt0 + t) * 16 + mr] =
                    (__bf16)acc[t][r];
    }
}

// ---------------------------------------------------------------- out proj
// Same 32x64-per-wave blocking as the QKV GEMM.
__global__ __launch_bounds__(256) void outproj_kernel(const __bf16* __restrict__ ctxb,
                                                      const __bf16* __restrict__ wo,
                                                      const float*  __restrict__ bias,
                                                      float* __restrict__ out) {
    const int wave = threadIdx.x >> 5;
    const int lane = threadIdx.x & 31;
    const int half = lane >> 4;
    const int mr   = lane & 15;
    const int m0 = blockIdx.x * 32;
    const int n0 = (blockIdx.y * 8 + wave) * 64;

    const __bf16* arow0 = ctxb + (size_t)(m0 + mr) * HQ;
    const __bf16* arow1 = ctxb + (size_t)(m0 + 16 + mr) * HQ;
    const __bf16* brow  = wo + (size_t)(n0 + mr) * HQ;

    f32x8 acc[2][4];
#pragma unroll
    for (int mi = 0; mi < 2; ++mi)
#pragma unroll
        for (int t = 0; t < 4; ++t) acc[mi][t] = zero8();

    for (int k0 = 0; k0 < HQ; k0 += 32) {
        AFrag a0, a1;
        a0.p.lo = *(const bf16x8*)(arow0 + k0 + 16 * half);
        a0.p.hi = *(const bf16x8*)(arow0 + k0 + 16 * half + 8);
        a1.p.lo = *(const bf16x8*)(arow1 + k0 + 16 * half);
        a1.p.hi = *(const bf16x8*)(arow1 + k0 + 16 * half + 8);
#pragma unroll
        for (int t = 0; t < 4; ++t) {
            bf16x16 bfr = *(const bf16x16*)(brow + (size_t)t * 16 * HQ + k0 + 16 * half);
            acc[0][t] = wmma_bf16(a0.v, bfr, acc[0][t]);
            acc[1][t] = wmma_bf16(a1.v, bfr, acc[1][t]);
        }
    }

#pragma unroll
    for (int t = 0; t < 4; ++t) {
        const int n = n0 + t * 16 + mr;
        const float bv = bias[n];
#pragma unroll
        for (int mi = 0; mi < 2; ++mi)
#pragma unroll
            for (int r = 0; r < 8; ++r) {
                const int m = m0 + 16 * mi + 8 * half + r;
                out[(size_t)m * HQ + n] = acc[mi][t][r] + bv;
            }
    }
}

// ---------------------------------------------------------------- launch
extern "C" void kernel_launch(void* const* d_in, const int* in_sizes, int n_in,
                              void* d_out, int out_size, void* d_ws, size_t ws_size,
                              hipStream_t stream) {
    (void)in_sizes; (void)n_in; (void)out_size; (void)ws_size;
    const float* x      = (const float*)d_in[0];  // [B,S,H]
    const float* qkv_w  = (const float*)d_in[1];  // [3H,H]
    const float* qkv_b  = (const float*)d_in[2];  // [3H]
    const float* out_w  = (const float*)d_in[3];  // [H,H]
    const float* out_b  = (const float*)d_in[4];  // [H]
    float* out = (float*)d_out;

    char* ws = (char*)d_ws;
    size_t off = 0;
    auto take = [&](size_t bytes) -> char* {
        char* p = ws + off;
        off += (bytes + 255) & ~(size_t)255;
        return p;
    };
    __bf16* xb   = (__bf16*)take((size_t)BQ * SQ * HQ * 2);
    __bf16* qb   = (__bf16*)take((size_t)BQ * SQ * HQ * 2);
    __bf16* kb   = (__bf16*)take((size_t)BQ * SQ * HQ * 2);
    __bf16* vtb  = (__bf16*)take((size_t)BQ * HQ * SQ * 2);  // [B][H][S]
    __bf16* ctxb = (__bf16*)take((size_t)BQ * SQ * HQ * 2);
    __bf16* wqb  = (__bf16*)take((size_t)3 * HQ * HQ * 2);
    __bf16* wob  = (__bf16*)take((size_t)HQ * HQ * 2);

    const int nX = BQ * SQ * HQ;
    const int nW = 3 * HQ * HQ;
    const int nO = HQ * HQ;
    cvt_f32_to_bf16<<<(nX + 255) / 256, 256, 0, stream>>>(x, xb, nX);
    cvt_f32_to_bf16<<<(nW + 255) / 256, 256, 0, stream>>>(qkv_w, wqb, nW);
    cvt_f32_to_bf16<<<(nO + 255) / 256, 256, 0, stream>>>(out_w, wob, nO);

    dim3 gq(BQ * SQ / 32, (3 * HQ) / (8 * 64));   // (256, 6)
    qkv_gemm_kernel<<<gq, 256, 0, stream>>>(xb, wqb, qkv_b, qb, kb, vtb);

    const size_t smem = (size_t)16 * SQ * 4 + (size_t)16 * HQ * 2;  // 288 KB < 320 KB
    attn_kernel<<<BQ * (SQ / 16), 256, smem, stream>>>(qb, kb, vtb, ctxb);

    dim3 go(BQ * SQ / 32, HQ / (8 * 64));         // (256, 2)
    outproj_kernel<<<go, 256, 0, stream>>>(ctxb, wob, out_b, out);
}
